// CausalSelfAttention_57174604644983
// MI455X (gfx1250) — matrix operand order repaired and checked by reading, not compiled, not run on hardware
//
#include <hip/hip_runtime.h>

// ---------------------------------------------------------------------------
// Causal bilinear self-attention for MI455X (gfx1250, wave32, WMMA + TDM).
//   Q[b,h] = h[b] @ A[h]                  (bf16 WMMA GEMM, TDM-fed)
//   O[b,h] = softmax_causal(Q Kᵀ) V,  K = V = h   (flash attention, TDM-fed)
// ~250 GFLOP of matrix math vs ~160 MB of HBM traffic -> matrix-core bound;
// bf16 WMMA with fp32 accumulation. All WMMA operands except the (L2-cached,
// one-step-prefetched) Q fragments come from LDS tiles staged by the Tensor
// Data Mover, double-buffered so the DMA overlaps compute.
// Sizes fixed by the reference: B=4, T=2048, D=512, H=8.
// Workspace layout (needs 84 MiB):
//   [0,8M)    hb  : bf16 h      [B][T][D]
//   [8M,16M)  ht  : bf16 hᵀ     [B][D][T]
//   [16M,20M) At  : bf16 Aᵀ     [H][D(e)][D(d)]
//   [20M,84M) Qw  : bf16 Q      [B][H][T][D]
// ---------------------------------------------------------------------------

#define DEV static __device__ __forceinline__

typedef __attribute__((ext_vector_type(16))) __bf16 bf16x16;
typedef __attribute__((ext_vector_type(8)))  __bf16 bf16x8;
typedef __attribute__((ext_vector_type(8)))  float  f32x8;
typedef __attribute__((ext_vector_type(4)))  unsigned int v4u;
typedef __attribute__((ext_vector_type(8)))  int v8i;
typedef __attribute__((ext_vector_type(4)))  int v4i;

constexpr int B_ = 4;
constexpr int T_ = 2048;
constexpr int D_ = 512;
constexpr int H_ = 8;

#if defined(__has_builtin)
#  if __has_builtin(__builtin_amdgcn_tensor_load_to_lds) && \
      __has_builtin(__builtin_amdgcn_s_wait_tensorcnt)
#    define HAVE_TDM 1
#  else
#    define HAVE_TDM 0
#  endif
#else
#  define HAVE_TDM 0
#endif

DEV __bf16 f2bf(float f) {
  union { float f; unsigned u; } v; v.f = f;
  unsigned r = v.u + 0x7FFFu + ((v.u >> 16) & 1u);   // round-to-nearest-even
  unsigned short s = (unsigned short)(r >> 16);
  __bf16 o; __builtin_memcpy(&o, &s, 2);
  return o;
}

// A-fragment (16x32 bf16, MxK) from [m][k]-contiguous storage:
// lane holds row m = lane%16; lanes 0-15: elems0-15=K0..15;
// lanes 16-31: elems0-15=K16..31 (matches contiguous B fragments).
DEV bf16x16 load_afrag(const __bf16* rowk, int lane) {
  const int alo = (lane >> 4) << 4;
  bf16x8 lo = *(const bf16x8*)(rowk + alo);
  bf16x8 hi = *(const bf16x8*)(rowk + alo + 8);
  return __builtin_shufflevector(lo, hi, 0,1,2,3,4,5,6,7,8,9,10,11,12,13,14,15);
}

DEV float half16_max(float x) {
  x = fmaxf(x, __shfl_xor(x, 1, 32));
  x = fmaxf(x, __shfl_xor(x, 2, 32));
  x = fmaxf(x, __shfl_xor(x, 4, 32));
  x = fmaxf(x, __shfl_xor(x, 8, 32));
  return x;
}
DEV float half16_sum(float x) {
  x += __shfl_xor(x, 1, 32);
  x += __shfl_xor(x, 2, 32);
  x += __shfl_xor(x, 4, 32);
  x += __shfl_xor(x, 8, 32);
  return x;
}

DEV f32x8 wmma_bf16(bf16x16 a, bf16x16 b, f32x8 c) {
  return __builtin_amdgcn_wmma_f32_16x16x32_bf16(false, a, false, b,
                                                 (short)0, c, false, false);
}

// --- Tensor Data Mover: 2D bf16 tile -> LDS (D# built per ISA 08, §8.3/8.4).
// cfg packs data_size=2B, pad_enable, pad_interval (8B*2^c per row),
// pad_amount (4 DWORDs) so LDS row strides are 16B-aligned and conflict-free.
constexpr unsigned CFG_ROW1024 = (1u << 16) | (1u << 20) | (7u << 22) | (3u << 25);
constexpr unsigned CFG_ROW64   = (1u << 16) | (1u << 20) | (3u << 22) | (3u << 25);

DEV void tdm_load_2d(unsigned lds_off, const void* gptr, unsigned dim0,
                     unsigned dim1, unsigned tile0, unsigned tile1,
                     unsigned stride0, unsigned cfg) {
#if HAVE_TDM
  unsigned long long ga = (unsigned long long)(uintptr_t)gptr;
  v4u g0 = { 1u,                                   // count=1 valid descriptor
             lds_off,                              // lds_addr (bytes)
             (unsigned)ga,                         // global_addr[31:0]
             (unsigned)((ga >> 32) & 0x01FFFFFFull) | (2u << 30) }; // [56:32]|type=2
  v8i g1 = { (int)cfg,
             (int)((dim0 & 0xFFFFu) << 16),                          // dim0 lo
             (int)((dim0 >> 16) | ((dim1 & 0xFFFFu) << 16)),         // dim0 hi|dim1 lo
             (int)((dim1 >> 16) | ((tile0 & 0xFFFFu) << 16)),        // dim1 hi|tile0
             (int)(tile1 & 0xFFFFu),                                 // tile1, tile2=0
             (int)stride0, 0, 0 };                                   // dim0 stride
  v4i z4 = {0, 0, 0, 0};
#  if __clang_major__ >= 23
  v8i z8 = {0, 0, 0, 0, 0, 0, 0, 0};
  __builtin_amdgcn_tensor_load_to_lds(g0, g1, z4, z4, z8, 0);
#  else
  __builtin_amdgcn_tensor_load_to_lds(g0, g1, z4, z4, 0);
#  endif
#else
  (void)lds_off; (void)gptr; (void)dim0; (void)dim1;
  (void)tile0; (void)tile1; (void)stride0; (void)cfg;
#endif
}

// --- prep: h(fp32) -> hb(bf16) and ht(bf16 transpose) ----------------------
__global__ void k_prep_h(const float* __restrict__ h,
                         __bf16* __restrict__ hb, __bf16* __restrict__ ht) {
  int i = blockIdx.x * blockDim.x + threadIdx.x;
  if (i >= B_ * T_ * D_) return;
  int b   = i / (T_ * D_);
  int rem = i - b * (T_ * D_);
  int t   = rem / D_;
  int d   = rem - t * D_;
  __bf16 x = f2bf(h[i]);
  hb[i] = x;
  ht[((size_t)b * D_ + d) * T_ + t] = x;
}

// --- prep: A(fp32 [h][d][e]) -> At(bf16 [h][e][d]) --------------------------
__global__ void k_prep_A(const float* __restrict__ A, __bf16* __restrict__ At) {
  int i = blockIdx.x * blockDim.x + threadIdx.x;
  if (i >= H_ * D_ * D_) return;
  int hd  = i / (D_ * D_);
  int rem = i - hd * (D_ * D_);
  int e   = rem / D_;
  int d   = rem - e * D_;
  At[i] = f2bf(A[((size_t)hd * D_ + d) * D_ + e]);
}

// --- Q GEMM: Qw[b,h,t,e] = sum_d hb[b,t,d] * At[h,e,d] ----------------------
// WG = (b, head, 128-row t-block); waves = 4 t-subtiles x 2 e-halves, each
// 16 rows x 256 cols (16 f32x8 accumulators). Per 32-wide k-step the TDM
// stages At k-slice (512x32) and hb k-slice (128x32) into padded LDS
// (row stride 40 bf16 = 80 B, 16B-aligned, conflict-free), double-buffered.
// Dynamic LDS: At 2x40960 @0 | hb 2x10240 @81920 -> 102400 B.
__global__ void __launch_bounds__(256)
k_qgemm(const __bf16* __restrict__ hb, const __bf16* __restrict__ At,
        __bf16* __restrict__ Qw) {
  extern __shared__ __align__(16) unsigned char smem[];

  const int lane  = threadIdx.x & 31;
  const int wslot = threadIdx.x >> 5;
  const int tsub  = wslot & 3;
  const int eh    = wslot >> 2;
  const int tb    = blockIdx.x & 15;
  const int head  = (blockIdx.x >> 4) & 7;
  const int b     = blockIdx.x >> 7;
  const int t0    = tb * 128 + tsub * 16;
  const int e0    = eh * 256;
  const int ln    = lane & 15;
  const int klo   = (lane >> 4) << 4;

  const __bf16* ag = At + (size_t)head * D_ * D_;           // [e][d]
  const __bf16* hg = hb + ((size_t)b * T_ + tb * 128) * D_; // [t][d]

  f32x8 acc[16];
#pragma unroll
  for (int f = 0; f < 16; ++f) acc[f] = f32x8{};

#if HAVE_TDM
  if (wslot == 0) {  // prologue: stage k-step 0
    tdm_load_2d(0u,     ag, D_, D_,             32, D_,  D_, CFG_ROW64);
    tdm_load_2d(81920u, hg, D_, T_ - tb * 128,  32, 128, D_, CFG_ROW64);
  }
#else
  for (int idx = threadIdx.x; idx < 512 * 2; idx += 256) {
    int row = idx >> 1, seg = (idx & 1) << 4;
    *(bf16x8*)(smem + (row * 40 + seg) * 2) =
        *(const bf16x8*)(ag + (size_t)row * D_ + seg);
  }
  for (int idx = threadIdx.x; idx < 128 * 2; idx += 256) {
    int row = idx >> 1, seg = (idx & 1) << 4;
    *(bf16x8*)(smem + 81920 + (row * 40 + seg) * 2) =
        *(const bf16x8*)(hg + (size_t)row * D_ + seg);
  }
#endif

  for (int ks = 0; ks < 16; ++ks) {
    const int cur = ks & 1;
#if HAVE_TDM
    if (wslot == 0) __builtin_amdgcn_s_wait_tensorcnt(0);
#endif
    __syncthreads();  // buf[cur] ready; everyone done with buf[cur^1]
    if (ks + 1 < 16) {
      const int nxt = cur ^ 1;
      const int kn = (ks + 1) << 5;
#if HAVE_TDM
      if (wslot == 0) {
        tdm_load_2d((unsigned)nxt * 40960u, ag + kn, D_ - kn, D_, 32, D_, D_,
                    CFG_ROW64);
        tdm_load_2d(81920u + (unsigned)nxt * 10240u, hg + kn, D_ - kn,
                    T_ - tb * 128, 32, 128, D_, CFG_ROW64);
      }
#else
      __bf16* An = (__bf16*)(smem + nxt * 40960);
      __bf16* Hn = (__bf16*)(smem + 81920 + nxt * 10240);
      for (int idx = threadIdx.x; idx < 512 * 2; idx += 256) {
        int row = idx >> 1, seg = (idx & 1) << 4;
        *(bf16x8*)(An + row * 40 + seg) =
            *(const bf16x8*)(ag + (size_t)row * D_ + kn + seg);
      }
      for (int idx = threadIdx.x; idx < 128 * 2; idx += 256) {
        int row = idx >> 1, seg = (idx & 1) << 4;
        *(bf16x8*)(Hn + row * 40 + seg) =
            *(const bf16x8*)(hg + (size_t)row * D_ + kn + seg);
      }
#endif
    }
    const __bf16* Ac = (const __bf16*)(smem + cur * 40960);
    const __bf16* Hc = (const __bf16*)(smem + 81920 + cur * 10240);
    bf16x16 a = load_afrag(Hc + (tsub * 16 + ln) * 40, lane);
#pragma unroll
    for (int f = 0; f < 16; ++f) {
      bf16x16 bb = *(const bf16x16*)(Ac + (e0 + 16 * f + ln) * 40 + klo);
      acc[f] = wmma_bf16(a, bb, acc[f]);
    }
  }

  __bf16* qrow = Qw + (((size_t)b * H_ + head) * T_ + t0) * D_ + e0;
  const int mh = (lane >> 4) << 3;
#pragma unroll
  for (int f = 0; f < 16; ++f)
#pragma unroll
    for (int r = 0; r < 8; ++r)
      qrow[(size_t)(mh + r) * D_ + 16 * f + ln] = f2bf(acc[f][r]);
}

// --- flash attention --------------------------------------------------------
// WG = 8 waves, one (b, head, d-half), 8 consecutive 16-row t-tiles.
// Per causal 32-wide s-block: TDM stages K (32x512, LDS stride 520) and
// Vᵀ (256x32, LDS stride 40), double-buffered so DMA overlaps compute.
// S = Q Kᵀ (Q A-frags prefetched one k-step ahead from L2), online softmax via
// shfl row-reductions, P relayout through per-wave LDS, O += P V from LDS.
// Dynamic LDS: K 2x33280 @0 | Vt 2x20480 @66560 | P 8x16x40 @107520 -> 117760 B.
__global__ void __launch_bounds__(256)
k_attn(const __bf16* __restrict__ Qw, const __bf16* __restrict__ hb,
       const __bf16* __restrict__ ht, float* __restrict__ out) {
  extern __shared__ __align__(16) unsigned char smem[];

  const int lane  = threadIdx.x & 31;
  const int wslot = threadIdx.x >> 5;
  const int wid   = (blockIdx.x * blockDim.x + threadIdx.x) >> 5;
  const int trow  = wid & 127;
  const int dh    = (wid >> 7) & 1;
  const int head  = (wid >> 8) & 7;
  const int b     = wid >> 11;
  const int t0  = trow << 4;
  const int d0  = dh << 8;
  const int ln  = lane & 15;
  const int lh  = lane >> 4;
  const int klo = lh << 4;

  const __bf16* qrow  = Qw + (((size_t)b * H_ + head) * T_ + t0 + ln) * D_;
  const __bf16* kbase = hb + (size_t)b * T_ * D_;
  const __bf16* vbase = ht + ((size_t)b * D_ + d0) * T_;
  __bf16* Pw = (__bf16*)(smem + 107520) + wslot * (16 * 40);

  f32x8 acc[16];
#pragma unroll
  for (int f = 0; f < 16; ++f) acc[f] = f32x8{};
  float mrun[8], lrun[8];
#pragma unroll
  for (int r = 0; r < 8; ++r) { mrun[r] = -1e30f; lrun[r] = 0.0f; }

  const int nb    = (t0 >> 5) + 1;                 // this wave's causal blocks
  const int trb   = trow & ~7;                     // WG's first t-tile
  const int nbmax = ((trb + 7) >> 1) + 1;          // WG-uniform block count

#if HAVE_TDM
  if (wslot == 0) {  // prologue: stage s-block 0
    tdm_load_2d(0u,     kbase, D_, T_,  D_, 32,  D_, CFG_ROW1024);
    tdm_load_2d(66560u, vbase, T_, 256, 32, 256, T_, CFG_ROW64);
  }
#else
  for (int idx = threadIdx.x; idx < 32 * 32; idx += 256) {
    int row = idx >> 5, seg = (idx & 31) << 4;
    *(bf16x8*)(smem + (row * 520 + seg) * 2) =
        *(const bf16x8*)(kbase + (size_t)row * D_ + seg);
  }
  for (int idx = threadIdx.x; idx < 256 * 2; idx += 256) {
    int row = idx >> 1, seg = (idx & 1) << 4;
    *(bf16x8*)(smem + 66560 + (row * 40 + seg) * 2) =
        *(const bf16x8*)(vbase + (size_t)row * T_ + seg);
  }
#endif

  for (int blk = 0; blk < nbmax; ++blk) {
    const int cur = blk & 1;
#if HAVE_TDM
    if (wslot == 0) __builtin_amdgcn_s_wait_tensorcnt(0);
#endif
    __syncthreads();  // buf[cur] ready; everyone done with buf[cur^1]
    if (blk + 1 < nbmax) {
      const int nxt = cur ^ 1;
      const int s1 = (blk + 1) << 5;
#if HAVE_TDM
      if (wslot == 0) {
        tdm_load_2d((unsigned)nxt * 33280u, kbase + (size_t)s1 * D_, D_,
                    T_ - s1, D_, 32, D_, CFG_ROW1024);
        tdm_load_2d(66560u + (unsigned)nxt * 20480u, vbase + s1, T_ - s1, 256,
                    32, 256, T_, CFG_ROW64);
      }
#else
      __bf16* Kn = (__bf16*)(smem + nxt * 33280);
      __bf16* Vn = (__bf16*)(smem + 66560 + nxt * 20480);
      for (int idx = threadIdx.x; idx < 32 * 32; idx += 256) {
        int row = idx >> 5, seg = (idx & 31) << 4;
        *(bf16x8*)(Kn + row * 520 + seg) =
            *(const bf16x8*)(kbase + (size_t)(s1 + row) * D_ + seg);
      }
      for (int idx = threadIdx.x; idx < 256 * 2; idx += 256) {
        int row = idx >> 1, seg = (idx & 1) << 4;
        *(bf16x8*)(Vn + row * 40 + seg) =
            *(const bf16x8*)(vbase + (size_t)row * T_ + s1 + seg);
      }
#endif
    }

    if (blk < nb) {
      const int s0 = blk << 5;
      const __bf16* Ks = (const __bf16*)(smem + cur * 33280);
      const __bf16* Vs = (const __bf16*)(smem + 66560 + cur * 20480);

      // ---- S = Q Kᵀ (16x32), Q prefetched one k-step ahead ----
      f32x8 sa = {}, sb = {};
      bf16x16 qa = load_afrag(qrow, lane);
#pragma unroll
      for (int kc = 0; kc < D_; kc += 32) {
        bf16x16 qn = qa;
        if (kc + 32 < D_) qn = load_afrag(qrow + kc + 32, lane);
        bf16x16 kb0 = *(const bf16x16*)(Ks + (ln) * 520 + kc + klo);
        bf16x16 kb1 = *(const bf16x16*)(Ks + (16 + ln) * 520 + kc + klo);
        sa = wmma_bf16(qa, kb0, sa);
        sb = wmma_bf16(qa, kb1, sb);
        qa = qn;
      }

      if (s0 + 31 > t0) {  // diagonal block: causal mask
#pragma unroll
        for (int r = 0; r < 8; ++r) {
          const int t = t0 + r + (lh << 3);
          if (s0 + ln > t)      sa[r] = -1e30f;
          if (s0 + 16 + ln > t) sb[r] = -1e30f;
        }
      }

      // ---- online softmax (rows live across 16-lane halves) ----
      float scale[8];
#pragma unroll
      for (int r = 0; r < 8; ++r) {
        float tm = half16_max(fmaxf(sa[r], sb[r]));
        float mn = fmaxf(mrun[r], tm);
        scale[r] = __expf(mrun[r] - mn);
        mrun[r]  = mn;
      }
#pragma unroll
      for (int r = 0; r < 8; ++r) {
        float p0 = __expf(sa[r] - mrun[r]);
        float p1 = __expf(sb[r] - mrun[r]);
        Pw[(r + (lh << 3)) * 40 + ln]      = f2bf(p0);
        Pw[(r + (lh << 3)) * 40 + 16 + ln] = f2bf(p1);
        lrun[r] = lrun[r] * scale[r] + half16_sum(p0 + p1);
      }
#pragma unroll
      for (int f = 0; f < 16; ++f)
#pragma unroll
        for (int r = 0; r < 8; ++r) acc[f][r] *= scale[r];

      // same-wave LDS round trip: DS ops are in-order per wave, no barrier
      bf16x16 pa = load_afrag(Pw + ln * 40, lane);
#pragma unroll
      for (int f = 0; f < 16; ++f) {
        bf16x16 vb = *(const bf16x16*)(Vs + (16 * f + ln) * 40 + klo);
        acc[f] = wmma_bf16(pa, vb, acc[f]);
      }
    }
  }

  float inv[8];
#pragma unroll
  for (int r = 0; r < 8; ++r) inv[r] = 1.0f / lrun[r];
  float* orow = out + ((size_t)b * T_ + t0) * (H_ * D_) + head * D_ + d0;
  const int mh = lh << 3;
#pragma unroll
  for (int f = 0; f < 16; ++f)
#pragma unroll
    for (int r = 0; r < 8; ++r)
      orow[(size_t)(mh + r) * (H_ * D_) + 16 * f + ln] = acc[f][r] * inv[r];
}

// ---------------------------------------------------------------------------
extern "C" void kernel_launch(void* const* d_in, const int* in_sizes, int n_in,
                              void* d_out, int out_size, void* d_ws,
                              size_t ws_size, hipStream_t stream) {
  (void)in_sizes; (void)n_in; (void)out_size; (void)ws_size;
  const float* h = (const float*)d_in[0];
  const float* A = (const float*)d_in[1];
  float* out = (float*)d_out;

  char* ws = (char*)d_ws;
  __bf16* hb = (__bf16*)(ws);
  __bf16* ht = (__bf16*)(ws + ((size_t)8 << 20));
  __bf16* At = (__bf16*)(ws + ((size_t)16 << 20));
  __bf16* Qw = (__bf16*)(ws + ((size_t)20 << 20));

  {
    int n = B_ * T_ * D_;
    k_prep_h<<<(n + 255) / 256, 256, 0, stream>>>(h, hb, ht);
  }
  {
    int n = H_ * D_ * D_;
    k_prep_A<<<(n + 255) / 256, 256, 0, stream>>>(A, At);
  }
  {
    int wgs = B_ * H_ * (T_ / 128);           // (b, head, 128-row t-block)
    k_qgemm<<<wgs, 256, 102400, stream>>>(hb, At, Qw);
  }
  {
    int waves = B_ * H_ * 2 * (T_ / 16);      // (b, head, d-half, t-tile)
    k_attn<<<waves * 32 / 256, 256, 117760, stream>>>(Qw, hb, ht, out);
  }
}
